// SlicingLinearBlock_12438225289884
// MI455X (gfx1250) — matrix operand order repaired and checked by reading, not compiled, not run on hardware
//
#include <hip/hip_runtime.h>

// SlicingLinearBlock fused GEMM for MI455X (gfx1250, wave32).
// out[16384,4096] = x[16384,256] @ W'[256,4096], W'[k][o] = W[k/8][o][k%8].
// bf16x3 split-precision WMMA (hi*hi + hi*lo + lo*hi, f32 accumulate).
// Pass 1 converts x/W to bf16 hi/lo planes in d_ws (L2-resident).
// Pass 2: pure-WMMA GEMM; tiles stream global->LDS via async-to-LDS (no
// data VGPRs, ASYNCcnt-tracked), double-buffered LDS, zero VALU in K loop.

typedef __attribute__((ext_vector_type(16))) __bf16 v16bf;
typedef __attribute__((ext_vector_type(8)))  float  v8f;
typedef __attribute__((ext_vector_type(4)))  int    v4i;

#define GEMM_M 16384
#define GEMM_N 4096
#define GEMM_K 256
#define BM 128
#define BN 128
#define BK 32
#define NT (GEMM_K / BK)
#define LDT 40   // padded LDS row stride (bf16 elems): 80B rows -> conflict-free frag reads

// ---- CDNA5 async global->LDS copy (no VGPR data; tracked by ASYNCcnt) ----
typedef __attribute__((address_space(1))) v4i gv4i;   // global int4
typedef __attribute__((address_space(3))) v4i lv4i;   // LDS int4

__device__ __forceinline__ void async_cp_b128(const void* g, void* l) {
#if defined(__AMDGCN__) && __has_builtin(__builtin_amdgcn_global_load_async_to_lds_b128)
  __builtin_amdgcn_global_load_async_to_lds_b128((gv4i*)g, (lv4i*)l, 0, 0);
#else
  unsigned lds = (unsigned)(unsigned long long)l;       // LDS aperture: addr[31:0] = LDS offset
  unsigned long long ga = (unsigned long long)g;
  asm volatile("global_load_async_to_lds_b128 %0, %1, off" :: "v"(lds), "v"(ga) : "memory");
#endif
}

__device__ __forceinline__ void wait_asynccnt_8() {
#if defined(__AMDGCN__) && __has_builtin(__builtin_amdgcn_s_wait_asynccnt)
  __builtin_amdgcn_s_wait_asynccnt(8);
#else
  asm volatile("s_wait_asynccnt 8" ::: "memory");
#endif
}
__device__ __forceinline__ void wait_asynccnt_0() {
#if defined(__AMDGCN__) && __has_builtin(__builtin_amdgcn_s_wait_asynccnt)
  __builtin_amdgcn_s_wait_asynccnt(0);
#else
  asm volatile("s_wait_asynccnt 0" ::: "memory");
#endif
}

__device__ __forceinline__ unsigned short f32_bf16_rne(float f) {
  union { float f; unsigned u; } v; v.f = f;
  unsigned r = v.u + 0x7FFFu + ((v.u >> 16) & 1u);
  return (unsigned short)(r >> 16);
}

__device__ __forceinline__ void split2(float f, unsigned short &hi, unsigned short &lo) {
  hi = f32_bf16_rne(f);
  union { unsigned u; float f; } hf; hf.u = ((unsigned)hi) << 16;
  lo = f32_bf16_rne(f - hf.f);
}

// A-fragment (16x32 bf16, ISA 7.12.2): lane r,g holds K = {g*8..g*8+7} U {16+g*8..16+g*8+7}
__device__ __forceinline__ v16bf frag_ld_a(const unsigned short* rowp, int g) {
  v16bf f;
  ((uint4*)&f)[0] = *(const uint4*)(rowp + g * 8);
  ((uint4*)&f)[1] = *(const uint4*)(rowp + 16 + g * 8);
  return f;
}

// B-fragment (32x16 bf16): lane n,g holds K = {g*8..g*8+7} U {16+g*8..16+g*8+7}
__device__ __forceinline__ v16bf frag_ld_b(const unsigned short* rowp, int g) {
  v16bf f;
  ((uint4*)&f)[0] = *(const uint4*)(rowp + g * 8);
  ((uint4*)&f)[1] = *(const uint4*)(rowp + 16 + g * 8);
  return f;
}

// ---------------------------------------------------------------------------
// Pass 1a: x[16384,256] f32 -> x_hi/x_lo [16384,256] bf16 planes
__global__ __launch_bounds__(256)
void slb_cvt_x(const float* __restrict__ x,
               unsigned short* __restrict__ xh, unsigned short* __restrict__ xl) {
  const size_t i = ((size_t)blockIdx.x * 256 + threadIdx.x) * 8;
  float tmp[8];
  ((float4*)tmp)[0] = ((const float4*)(x + i))[0];
  ((float4*)tmp)[1] = ((const float4*)(x + i))[1];
  union { unsigned short s[8]; uint4 v; } H, L;
  #pragma unroll
  for (int e = 0; e < 8; ++e) split2(tmp[e], H.s[e], L.s[e]);
  *(uint4*)(xh + i) = H.v;
  *(uint4*)(xl + i) = L.v;
}

// Pass 1b: W[32,4096,8] f32 -> Wt_hi/Wt_lo [4096,256] bf16, Wt[o][k]=W[k/8][o][k%8]
__global__ __launch_bounds__(256)
void slb_cvt_w(const float* __restrict__ W,
               unsigned short* __restrict__ wh, unsigned short* __restrict__ wl) {
  const int t = blockIdx.x * 256 + threadIdx.x;   // 0 .. 4096*32-1
  const int s = t & 31;                           // slice = k/8
  const int o = t >> 5;                           // output channel
  const float* pw = W + ((size_t)s * GEMM_N + o) * 8;
  float tmp[8];
  ((float4*)tmp)[0] = ((const float4*)pw)[0];
  ((float4*)tmp)[1] = ((const float4*)pw)[1];
  union { unsigned short s[8]; uint4 v; } H, L;
  #pragma unroll
  for (int e = 0; e < 8; ++e) split2(tmp[e], H.s[e], L.s[e]);
  *(uint4*)(wh + (size_t)o * GEMM_K + s * 8) = H.v;
  *(uint4*)(wl + (size_t)o * GEMM_K + s * 8) = L.v;
}

// ---------------------------------------------------------------------------
// Pass 2: pure-WMMA GEMM; async global->LDS double-buffered staging.
__global__ __launch_bounds__(256)
void slb_gemm_pre(const unsigned short* __restrict__ xh, const unsigned short* __restrict__ xl,
                  const unsigned short* __restrict__ wh, const unsigned short* __restrict__ wl,
                  float* __restrict__ out) {
  __shared__ unsigned short As_hi[2][BM][LDT];
  __shared__ unsigned short As_lo[2][BM][LDT];
  __shared__ unsigned short Bs_hi[2][BN][LDT];   // [n][k]: B frags are contiguous rows
  __shared__ unsigned short Bs_lo[2][BN][LDT];

  const int tid  = threadIdx.x;
  const int lane = tid & 31;
  const int wv   = tid >> 5;       // 8 waves
  const int wm   = wv & 3;         // 4 waves along M -> 32 rows each
  const int wn   = wv >> 2;        // 2 waves along N -> 64 cols each
  const int r    = lane & 15;
  const int g    = lane >> 4;

  const int m_block = blockIdx.y * BM;
  const int n_block = blockIdx.x * BN;

  // staging: 2 threads per tile row; each moves 16 bf16 (2 x async b128) per plane
  const int rowA = tid >> 1;
  const int segA = tid & 1;

  const unsigned short* pxh = xh + (size_t)(m_block + rowA) * GEMM_K + segA * 16;
  const unsigned short* pxl = xl + (size_t)(m_block + rowA) * GEMM_K + segA * 16;
  const unsigned short* pwh = wh + (size_t)(n_block + rowA) * GEMM_K + segA * 16;
  const unsigned short* pwl = wl + (size_t)(n_block + rowA) * GEMM_K + segA * 16;
  const int so = segA * 16;

  auto issue_tile = [&](int kt) {            // 8 async b128 per thread, in order
    const int b = kt & 1;
    const int o = kt * BK;
    async_cp_b128(pxh + o,     &As_hi[b][rowA][so]);
    async_cp_b128(pxh + o + 8, &As_hi[b][rowA][so + 8]);
    async_cp_b128(pxl + o,     &As_lo[b][rowA][so]);
    async_cp_b128(pxl + o + 8, &As_lo[b][rowA][so + 8]);
    async_cp_b128(pwh + o,     &Bs_hi[b][rowA][so]);
    async_cp_b128(pwh + o + 8, &Bs_hi[b][rowA][so + 8]);
    async_cp_b128(pwl + o,     &Bs_lo[b][rowA][so]);
    async_cp_b128(pwl + o + 8, &Bs_lo[b][rowA][so + 8]);
  };

  v8f acc[2][4];
  v8f zero = {0.f, 0.f, 0.f, 0.f, 0.f, 0.f, 0.f, 0.f};
  #pragma unroll
  for (int i = 0; i < 2; ++i)
    #pragma unroll
    for (int j = 0; j < 4; ++j) acc[i][j] = zero;

  issue_tile(0);

  #pragma unroll 1
  for (int kt = 0; kt < NT; ++kt) {
    if (kt < NT - 1) {
      issue_tile(kt + 1);     // prefetch into the other buffer
      wait_asynccnt_8();      // tile kt's 8 (oldest, in-order) async ops done
    } else {
      wait_asynccnt_0();
    }
    __syncthreads();          // tile kt visible to all waves

    const int b = kt & 1;
    v16bf a_hi[2], a_lo[2];
    #pragma unroll
    for (int mf = 0; mf < 2; ++mf) {
      const int row = wm * 32 + mf * 16 + r;
      a_hi[mf] = frag_ld_a(&As_hi[b][row][0], g);
      a_lo[mf] = frag_ld_a(&As_lo[b][row][0], g);
    }
    #pragma unroll
    for (int nf = 0; nf < 4; ++nf) {
      const int col = wn * 64 + nf * 16 + r;
      v16bf b_hi = frag_ld_b(&Bs_hi[b][col][0], g);
      v16bf b_lo = frag_ld_b(&Bs_lo[b][col][0], g);
      #pragma unroll
      for (int mf = 0; mf < 2; ++mf) {
        acc[mf][nf] = __builtin_amdgcn_wmma_f32_16x16x32_bf16(
            false, a_hi[mf], false, b_hi, (short)0, acc[mf][nf], false, false);
        acc[mf][nf] = __builtin_amdgcn_wmma_f32_16x16x32_bf16(
            false, a_hi[mf], false, b_lo, (short)0, acc[mf][nf], false, false);
        acc[mf][nf] = __builtin_amdgcn_wmma_f32_16x16x32_bf16(
            false, a_lo[mf], false, b_hi, (short)0, acc[mf][nf], false, false);
      }
    }
    __syncthreads();          // all reads of buffer b done before it is re-filled
  }

  // Epilogue: C layout (ISA 7.12.2): VGPR v -> row v + g*8, lane (r) -> col
  #pragma unroll
  for (int mf = 0; mf < 2; ++mf) {
    #pragma unroll
    for (int nf = 0; nf < 4; ++nf) {
      const int row0 = m_block + wm * 32 + mf * 16 + g * 8;
      const int col  = n_block + wn * 64 + nf * 16 + r;
      float* po = out + (size_t)row0 * GEMM_N + col;
      #pragma unroll
      for (int v = 0; v < 8; ++v) po[(size_t)v * GEMM_N] = acc[mf][nf][v];
    }
  }
}

// ---------------------------------------------------------------------------
// Fallback (ws too small): fused conversion GEMM.
__global__ __launch_bounds__(256)
void slb_gemm_fused(const float* __restrict__ x, const float* __restrict__ W,
                    float* __restrict__ out) {
  __shared__ unsigned short As_hi[BM][LDT];
  __shared__ unsigned short As_lo[BM][LDT];
  __shared__ unsigned short Bs_hi[BN][LDT];
  __shared__ unsigned short Bs_lo[BN][LDT];

  const int tid  = threadIdx.x;
  const int lane = tid & 31;
  const int wv   = tid >> 5;
  const int wm   = wv & 3;
  const int wn   = wv >> 2;
  const int r    = lane & 15;
  const int g    = lane >> 4;

  const int m_block = blockIdx.y * BM;
  const int n_block = blockIdx.x * BN;

  const int rowA  = tid >> 1;
  const int segA  = tid & 1;
  const int halfB = tid & 1;

  float4 aR[4], bR[4];

  auto global_load_tiles = [&](int kt) {
    const float* pa = x + (size_t)(m_block + rowA) * GEMM_K + kt * BK + segA * 16;
    #pragma unroll
    for (int i = 0; i < 4; ++i) aR[i] = ((const float4*)pa)[i];
    const int sbase = kt * 4 + halfB * 2;
    #pragma unroll
    for (int j = 0; j < 2; ++j) {
      const float* pb = W + ((size_t)(sbase + j) * GEMM_N + (n_block + rowA)) * 8;
      bR[2 * j + 0] = ((const float4*)pb)[0];
      bR[2 * j + 1] = ((const float4*)pb)[1];
    }
  };

  auto lds_store_tiles = [&]() {
    const float* af = (const float*)aR;
    const float* bf = (const float*)bR;
    #pragma unroll
    for (int e = 0; e < 16; e += 2) {
      unsigned short h0, l0, h1, l1;
      split2(af[e], h0, l0);
      split2(af[e + 1], h1, l1);
      *(unsigned*)&As_hi[rowA][segA * 16 + e] = (unsigned)h0 | ((unsigned)h1 << 16);
      *(unsigned*)&As_lo[rowA][segA * 16 + e] = (unsigned)l0 | ((unsigned)l1 << 16);
    }
    #pragma unroll
    for (int e = 0; e < 16; e += 2) {
      unsigned short h0, l0, h1, l1;
      split2(bf[e], h0, l0);
      split2(bf[e + 1], h1, l1);
      *(unsigned*)&Bs_hi[rowA][halfB * 16 + e] = (unsigned)h0 | ((unsigned)h1 << 16);
      *(unsigned*)&Bs_lo[rowA][halfB * 16 + e] = (unsigned)l0 | ((unsigned)l1 << 16);
    }
  };

  v8f acc[2][4];
  v8f zero = {0.f, 0.f, 0.f, 0.f, 0.f, 0.f, 0.f, 0.f};
  #pragma unroll
  for (int i = 0; i < 2; ++i)
    #pragma unroll
    for (int j = 0; j < 4; ++j) acc[i][j] = zero;

  global_load_tiles(0);

  #pragma unroll 1
  for (int kt = 0; kt < NT; ++kt) {
    lds_store_tiles();
    __syncthreads();
    if (kt < NT - 1) global_load_tiles(kt + 1);

    v16bf a_hi[2], a_lo[2];
    #pragma unroll
    for (int mf = 0; mf < 2; ++mf) {
      const int row = wm * 32 + mf * 16 + r;
      a_hi[mf] = frag_ld_a(&As_hi[row][0], g);
      a_lo[mf] = frag_ld_a(&As_lo[row][0], g);
    }
    #pragma unroll
    for (int nf = 0; nf < 4; ++nf) {
      const int col = wn * 64 + nf * 16 + r;
      v16bf b_hi = frag_ld_b(&Bs_hi[col][0], g);
      v16bf b_lo = frag_ld_b(&Bs_lo[col][0], g);
      #pragma unroll
      for (int mf = 0; mf < 2; ++mf) {
        acc[mf][nf] = __builtin_amdgcn_wmma_f32_16x16x32_bf16(
            false, a_hi[mf], false, b_hi, (short)0, acc[mf][nf], false, false);
        acc[mf][nf] = __builtin_amdgcn_wmma_f32_16x16x32_bf16(
            false, a_hi[mf], false, b_lo, (short)0, acc[mf][nf], false, false);
        acc[mf][nf] = __builtin_amdgcn_wmma_f32_16x16x32_bf16(
            false, a_lo[mf], false, b_hi, (short)0, acc[mf][nf], false, false);
      }
    }
    __syncthreads();
  }

  #pragma unroll
  for (int mf = 0; mf < 2; ++mf) {
    #pragma unroll
    for (int nf = 0; nf < 4; ++nf) {
      const int row0 = m_block + wm * 32 + mf * 16 + g * 8;
      const int col  = n_block + wn * 64 + nf * 16 + r;
      float* po = out + (size_t)row0 * GEMM_N + col;
      #pragma unroll
      for (int v = 0; v < 8; ++v) po[(size_t)v * GEMM_N] = acc[mf][nf][v];
    }
  }
}

// ---------------------------------------------------------------------------
extern "C" void kernel_launch(void* const* d_in, const int* in_sizes, int n_in,
                              void* d_out, int out_size, void* d_ws, size_t ws_size,
                              hipStream_t stream) {
  const float* x = (const float*)d_in[0];   // [16384, 256] f32
  const float* W = (const float*)d_in[1];   // [32, 4096, 8] f32
  float* out = (float*)d_out;               // [16384, 4096] f32

  const size_t nx = (size_t)GEMM_M * GEMM_K;        // 4,194,304
  const size_t nw = (size_t)GEMM_N * GEMM_K;        // 1,048,576
  const size_t need = (2 * nx + 2 * nw) * sizeof(unsigned short);   // 20 MB

  dim3 grid(GEMM_N / BN, GEMM_M / BM);              // (32, 128)

  if (ws_size >= need) {
    unsigned short* xh = (unsigned short*)d_ws;
    unsigned short* xl = xh + nx;
    unsigned short* wh = xl + nx;
    unsigned short* wl = wh + nw;
    slb_cvt_x<<<(int)(nx / (256 * 8)), 256, 0, stream>>>(x, xh, xl);
    slb_cvt_w<<<(int)(nw / (256 * 8)), 256, 0, stream>>>(W, wh, wl);
    slb_gemm_pre<<<grid, 256, 0, stream>>>(xh, xl, wh, wl, out);
  } else {
    slb_gemm_fused<<<grid, 256, 0, stream>>>(x, W, out);
  }
}
